// AdaptiveFourierNeuralOperator_40604620817191
// MI455X (gfx1250) — matrix operand order repaired and checked by reading, not compiled, not run on hardware
//
#include <hip/hip_runtime.h>
#include <hip/hip_bf16.h>
#include <math.h>
#include <stdint.h>
#include <stddef.h>

// ---------------------------------------------------------------------------
// AFNO block for MI455X (gfx1250, wave32, WMMA, TDM).
// All stages (rfft2/irfft2 recast as 64-pt DFT GEMMs, bias GEMM, block MLP)
// run on v_wmma_f32_16x16x32_f16 (f16 operands, f32 accumulate).
// ~95 GFLOP vs ~0.35 GB HBM -> f16 WMMA compute sits at the 23.3 TB/s floor.
// Weight/table fragments: contiguous global_load_b128 pairs. All strided
// data-side operands (dft_w, dft_h x2, idft_w) are staged into LDS by the
// Tensor Data Mover (tensor_load_to_lds + s_wait_tensorcnt) and consumed as
// LDS fragments, keeping the vector memory path on full-width accesses.
// ---------------------------------------------------------------------------

typedef __attribute__((ext_vector_type(16))) _Float16 v16h;
typedef __attribute__((ext_vector_type(8)))  _Float16 v8h;
typedef __attribute__((ext_vector_type(4)))  _Float16 v4h;
typedef __attribute__((ext_vector_type(8)))  float    v8f;
typedef __attribute__((ext_vector_type(4)))  unsigned int u32x4;
typedef __attribute__((ext_vector_type(4)))  int      i32x4v;
typedef __attribute__((ext_vector_type(8)))  int      i32x8v;

#define DEVINL static __device__ __forceinline__

#define BATCH 8
#define DIMC  768
#define HH    64
#define WWW   64
#define KWF   33               // W/2+1
#define NBLK  8
#define BSZ   96
#define NPOS  (BATCH*HH*KWF)   // 16896 frequency positions
#define LDFREQ (KWF*DIMC)      // 25344, row (h or kh) stride in Y/Z buffers

DEVINL v8f wmma16(v16h a, v16h b, v8f c) {
  return __builtin_amdgcn_wmma_f32_16x16x32_f16(
      false, a, false, b, (short)0, c, false, false);
}

// ---- A fragment (16x32 f16), fast path: per-lane two contiguous 16B loads.
// Lane L holds row m0+(L&15); kb = k0 + 16*(L>>4); elements 0-15 = K kb..kb+15
// (contiguous order, matching the B loaders).
DEVINL v16h load_aF(const _Float16* __restrict__ src, int ld, int m0, int k0) {
  int lane = threadIdx.x & 31;
  const _Float16* row = src + (size_t)(m0 + (lane & 15)) * ld + k0 + ((lane >> 4) << 4);
  v8h lo = *(const v8h*)row;
  v8h hi = *(const v8h*)(row + 8);
  v16h a;
#pragma unroll
  for (int e = 0; e < 8; ++e) { a[e] = lo[e]; a[8 + e] = hi[e]; }
  return a;
}

// ---- B fragment (32x16 f16) from N-major (transposed) storage: per-lane
// 16 contiguous K halves = two 16B loads. Lane L = column n0+(L&15),
// element e = K (k0 + 16*(L>>4) + e).
DEVINL v16h load_bT(const _Float16* __restrict__ srcT, int ld, int k0, int n0) {
  int lane = threadIdx.x & 31;
  const _Float16* row = srcT + (size_t)(n0 + (lane & 15)) * ld + k0 + ((lane >> 4) << 4);
  v8h lo = *(const v8h*)row;
  v8h hi = *(const v8h*)(row + 8);
  v16h b;
#pragma unroll
  for (int e = 0; e < 8; ++e) { b[e] = lo[e]; b[8 + e] = hi[e]; }
  return b;
}

// ---- B fragment from an LDS-staged strip laid out [K rows x 16 cols].
DEVINL v16h load_bLDS(const _Float16* strip, int k0) {
  int lane = threadIdx.x & 31;
  int nl = lane & 15;
  int kb = k0 + ((lane >> 4) << 4);
  v16h b;
#pragma unroll
  for (int e = 0; e < 16; ++e) b[e] = strip[(kb + e) * 16 + nl];
  return b;
}

DEVINL v16h load_bLDS_guard(const _Float16* strip, int k0, int Kb) {
  int lane = threadIdx.x & 31;
  int nl = lane & 15;
  int kb = k0 + ((lane >> 4) << 4);
  v16h b;
#pragma unroll
  for (int e = 0; e < 16; ++e) {
    int k = kb + e;
    b[e] = (k < Kb) ? strip[k * 16 + nl] : (_Float16)0.0f;
  }
  return b;
}

DEVINL float sshrink(float v) {
  return (v > 0.01f) ? (v - 0.01f) : ((v < -0.01f) ? (v + 0.01f) : 0.0f);
}

// ---------------------------------------------------------------------------
// Tensor Data Mover: 2D f16 tile (tile_x elems wide, rows deep, row stride in
// elems) from global -> LDS byte offset (static LDS block starts at 0).
// ---------------------------------------------------------------------------
#if __has_builtin(__builtin_amdgcn_tensor_load_to_lds)
#define HAVE_TDM 1
#else
#define HAVE_TDM 0
#endif

DEVINL void tdm_load_tile_f16(uint32_t lds_byte_off, const _Float16* gsrc,
                              uint32_t tile_x, uint32_t rows, uint32_t stride_elems) {
#if HAVE_TDM
  uint64_t ga = (uint64_t)(uintptr_t)gsrc;
  u32x4 g0;
  g0[0] = 1u;                                               // count=1, user flags 0
  g0[1] = lds_byte_off;                                     // lds_addr (bytes)
  g0[2] = (uint32_t)ga;                                     // global_addr[31:0]
  g0[3] = (uint32_t)((ga >> 32) & 0x01FFFFFFu) | (2u << 30);// addr[56:32], type=2
  const uint32_t td0 = 0x40000000u;                         // tensor_dim0: no OOB clip
  const uint32_t td1 = rows;                                // tensor_dim1
  i32x8v g1;
  g1[0] = (int)(1u << 16);                                  // mask=0, data_size=1 (2B)
  g1[1] = (int)((td0 & 0xFFFFu) << 16);                     // td0[15:0] @ bits 63:48
  g1[2] = (int)(((td0 >> 16) & 0xFFFFu) | ((td1 & 0xFFFFu) << 16));
  g1[3] = (int)(((td1 >> 16) & 0xFFFFu) | (tile_x << 16));  // tile_dim0 @ 127:112
  g1[4] = (int)(rows & 0xFFFFu);                            // tile_dim1, tile_dim2=0
  g1[5] = (int)stride_elems;                                // dim0_stride[31:0]
  g1[6] = 0;
  g1[7] = 0;
  i32x4v z4 = {0, 0, 0, 0};
#if defined(__clang_major__) && __clang_major__ >= 23
  i32x8v z8 = {0, 0, 0, 0, 0, 0, 0, 0};
  __builtin_amdgcn_tensor_load_to_lds(g0, g1, z4, z4, z8, 0);
#else
  __builtin_amdgcn_tensor_load_to_lds(g0, g1, z4, z4, 0);
#endif
#else
  (void)lds_byte_off; (void)gsrc; (void)tile_x; (void)rows; (void)stride_elems;
#endif
}

DEVINL void tdm_wait() {
#if __has_builtin(__builtin_amdgcn_s_wait_tensorcnt)
  __builtin_amdgcn_s_wait_tensorcnt(0);
#endif
  // The TDM writes LDS behind the compiler's back: force it to treat LDS as
  // modified so staged reads are not folded/ reordered.
  asm volatile("" ::: "memory");
}

// Manual fallback copy for toolchains without the TDM builtin.
DEVINL void manual_stage(_Float16* dst, const _Float16* src, int rows,
                         uint32_t stride_elems) {
  int lane = threadIdx.x & 31;
  for (int idx = lane; idx < rows * 16; idx += 32) {
    int k = idx >> 4, nn = idx & 15;
    dst[idx] = src[(size_t)k * stride_elems + nn];
  }
}

// ---------------------------------------------------------------------------
// Kernel A: x (f32) -> x_h (f16), float4-vectorized.
// ---------------------------------------------------------------------------
__global__ void __launch_bounds__(256) cvt_x_kernel(const float* __restrict__ x,
                                                    _Float16* __restrict__ xh) {
  int i = blockIdx.x * blockDim.x + threadIdx.x;     // quad index
  float4 v = ((const float4*)x)[i];
  v4h h;
  h[0] = (_Float16)v.x; h[1] = (_Float16)v.y;
  h[2] = (_Float16)v.z; h[3] = (_Float16)v.w;
  ((v4h*)xh)[i] = h;
}

// ---------------------------------------------------------------------------
// Kernel 0: f16 weights (N-major for B-fragment loads) + DFT basis tables.
// Ranges: Wb_h 589824 | w1rT 73728 | w1iT(+neg) 73728 | w2rT 73728 |
// w2iT(+neg) 73728 | CwT 3072 | ChT/iChT 4096 | D 4096  => 896000 threads
// ---------------------------------------------------------------------------
__global__ void __launch_bounds__(256) prep_kernel(
    const float* __restrict__ w1, const float* __restrict__ w2,
    const float* __restrict__ Wb,
    _Float16* Wb_h,
    _Float16* w1rT, _Float16* w1iT, _Float16* w1iNT,
    _Float16* w2rT, _Float16* w2iT, _Float16* w2iNT,
    _Float16* CwR, _Float16* CwI,
    _Float16* ChR, _Float16* ChI, _Float16* ChIN,
    _Float16* iChR, _Float16* iChI, _Float16* iChIN,
    _Float16* DrT, _Float16* DiT) {
  const float TW = 6.283185307179586f / 64.0f;
  int i = blockIdx.x * blockDim.x + threadIdx.x;
  if (i < DIMC * DIMC) {             // Wb is already N-major for out[m,d]=x.Wb[d,:]
    Wb_h[i] = (_Float16)Wb[i];
    return;
  }
  i -= DIMC * DIMC;
  const int WSZ = NBLK * BSZ * BSZ;  // 73728
  // transposed block weights: wT[nb][n][d] = w[nb][d][n]
  int nb = (i % WSZ) / (BSZ * BSZ), rm = (i % WSZ) % (BSZ * BSZ);
  int n = rm / BSZ, d = rm % BSZ;
  int srcT = nb * BSZ * BSZ + d * BSZ + n;
  if (i < WSZ) { w1rT[i] = (_Float16)w1[srcT]; return; }
  i -= WSZ;
  if (i < WSZ) { float v = w1[WSZ + srcT]; w1iT[i] = (_Float16)v; w1iNT[i] = (_Float16)(-v); return; }
  i -= WSZ;
  if (i < WSZ) { w2rT[i] = (_Float16)w2[srcT]; return; }
  i -= WSZ;
  if (i < WSZ) { float v = w2[WSZ + srcT]; w2iT[i] = (_Float16)v; w2iNT[i] = (_Float16)(-v); return; }
  i -= WSZ;
  if (i < 48 * 64) {                 // rfft-W basis, rows kw (pad 33..47 = 0)
    int kw = i / 64, w = i % 64;
    float c = 0.f, s = 0.f;
    if (kw < KWF) { float th = TW * (float)(w * kw); c = cosf(th) * 0.125f; s = -sinf(th) * 0.125f; }
    CwR[i] = (_Float16)c; CwI[i] = (_Float16)s;
    return;
  }
  i -= 48 * 64;
  if (i < 64 * 64) {                 // fft-H / ifft-H basis (symmetric index)
    int a = i / 64, b = i % 64;
    float th = TW * (float)(a * b);
    float c = cosf(th) * 0.125f, s = sinf(th) * 0.125f;
    ChR[i]  = (_Float16)c; ChI[i]  = (_Float16)(-s); ChIN[i]  = (_Float16)s;
    iChR[i] = (_Float16)c; iChI[i] = (_Float16)s;    iChIN[i] = (_Float16)(-s);
    return;
  }
  i -= 64 * 64;
  if (i < 64 * 64) {                 // irfft-W basis, cols k (pad 33..63 = 0)
    int w = i / 64, k = i % 64;
    float dr = 0.f, di = 0.f;
    if (k < KWF) {
      float wt = (k == 0 || k == 32) ? 1.0f : 2.0f;
      float th = TW * (float)(w * k);
      dr = wt * cosf(th) * 0.125f;
      di = -wt * sinf(th) * 0.125f;
    }
    DrT[i] = (_Float16)dr; DiT[i] = (_Float16)di;
  }
}

// ---------------------------------------------------------------------------
// Kernel 1: bias path out[m,d] = x[m,:].Wb[d,:] + bb[d]  (overwrites d_out).
// M=32768, N=768, K=768; A=x_h (fast), B=Wb_h (fast, N-major).
// ---------------------------------------------------------------------------
__global__ void __launch_bounds__(256) bias_gemm_kernel(
    const _Float16* __restrict__ xh, const _Float16* __restrict__ Wb_h,
    const float* __restrict__ bb, float* __restrict__ out) {
  int wv = (blockIdx.x * blockDim.x + threadIdx.x) >> 5;
  int lane = threadIdx.x & 31;
  int nt = wv % 48, mt = wv / 48;
  int m0 = mt * 16, n0 = nt * 16;
  v8f acc;
  float bv = bb[n0 + (lane & 15)];
#pragma unroll
  for (int r = 0; r < 8; ++r) acc[r] = bv;
  __builtin_prefetch(xh + (size_t)m0 * DIMC, 0, 1);
  for (int k0 = 0; k0 < DIMC; k0 += 32) {
    v16h a = load_aF(xh, DIMC, m0, k0);
    v16h b = load_bT(Wb_h, DIMC, k0, n0);
    acc = wmma16(a, b, acc);
  }
#pragma unroll
  for (int r = 0; r < 8; ++r) {
    int m = m0 + r + ((lane >> 4) << 3);
    out[(size_t)m * DIMC + n0 + (lane & 15)] = acc[r];
  }
}

// ---------------------------------------------------------------------------
// Kernel 2: rfft along W. Per (b,h): Y[kw,c] = CwT[kw,:].X[:,c]; f16 out.
// One workgroup per (bh, nt-octet); each wave TDM-stages its 64x16 X strip
// (row stride 768) into LDS, then runs 3 M-tiles from the staged copy.
// ---------------------------------------------------------------------------
__global__ void __launch_bounds__(256) dft_w_kernel(
    const _Float16* __restrict__ xh,
    const _Float16* __restrict__ CwR, const _Float16* __restrict__ CwI,
    _Float16* __restrict__ Yr, _Float16* __restrict__ Yi) {
  __shared__ _Float16 s_x[8][64 * 16];       // 16 KB
  int wvid = threadIdx.x >> 5;
  int lane = threadIdx.x & 31;
  int wg = blockIdx.x;
  int ng = wg % 6;
  int bh = wg / 6;                            // 0..511
  int n0 = (ng * 8 + wvid) * 16;
  const _Float16* X = xh + (size_t)bh * WWW * DIMC;
#if HAVE_TDM
  tdm_load_tile_f16((uint32_t)(wvid * 64 * 16 * sizeof(_Float16)),
                    X + n0, 16, 64, DIMC);
  tdm_wait();
#else
  manual_stage(&s_x[wvid][0], X + n0, 64, DIMC);
#endif
  const _Float16* sx = &s_x[wvid][0];
  size_t base = (size_t)bh * LDFREQ;
  for (int mt = 0; mt < 3; ++mt) {
    int m0 = mt * 16;
    v8f ar = {}, ai = {};
    for (int k0 = 0; k0 < 64; k0 += 32) {
      v16h fB  = load_bLDS(sx, k0);
      v16h fAr = load_aF(CwR, 64, m0, k0);
      v16h fAi = load_aF(CwI, 64, m0, k0);
      ar = wmma16(fAr, fB, ar);
      ai = wmma16(fAi, fB, ai);
    }
#pragma unroll
    for (int r = 0; r < 8; ++r) {
      int m = m0 + r + ((lane >> 4) << 3);   // kw
      if (m < KWF) {
        size_t o = base + (size_t)m * DIMC + n0 + (lane & 15);
        Yr[o] = (_Float16)ar[r];
        Yi[o] = (_Float16)ai[r];
      }
    }
  }
}

// ---------------------------------------------------------------------------
// Kernel 3/5: complex DFT along H (tables select forward/inverse):
//   Or = Ar@Br + AiN@Bi ; Oi = Ar@Bi + Ai@Br
// One workgroup per (b, kw, nt-octet); each wave TDM-stages its strided
// 64x16 B strips (r,i) into LDS, then runs all 4 M-tiles from the staged copy.
// ---------------------------------------------------------------------------
__global__ void __launch_bounds__(256) dft_h_kernel(
    const _Float16* __restrict__ Ar, const _Float16* __restrict__ Ai,
    const _Float16* __restrict__ AiN,
    const _Float16* __restrict__ Br, const _Float16* __restrict__ Bi,
    _Float16* __restrict__ Or, _Float16* __restrict__ Oi) {
  __shared__ _Float16 s_b[8][2][64 * 16];    // 32 KB
  int wvid = threadIdx.x >> 5;
  int lane = threadIdx.x & 31;
  int wg = blockIdx.x;
  int ng = wg % 6; wg /= 6;
  int kw = wg % KWF;
  int b  = wg / KWF;
  int n0 = (ng * 8 + wvid) * 16;
  size_t base = (size_t)b * (HH * LDFREQ) + (size_t)kw * DIMC;
#if HAVE_TDM
  uint32_t off_r = (uint32_t)(wvid * 2 * 64 * 16 * sizeof(_Float16));
  uint32_t off_i = off_r + 64 * 16 * sizeof(_Float16);
  tdm_load_tile_f16(off_r, Br + base + n0, 16, 64, LDFREQ);
  tdm_load_tile_f16(off_i, Bi + base + n0, 16, 64, LDFREQ);
  tdm_wait();
#else
  manual_stage(&s_b[wvid][0][0], Br + base + n0, 64, LDFREQ);
  manual_stage(&s_b[wvid][1][0], Bi + base + n0, 64, LDFREQ);
#endif
  const _Float16* sbr = &s_b[wvid][0][0];
  const _Float16* sbi = &s_b[wvid][1][0];
  for (int mt = 0; mt < 4; ++mt) {
    int m0 = mt * 16;
    v8f or_ = {}, oi_ = {};
    for (int k0 = 0; k0 < 64; k0 += 32) {
      v16h fAr  = load_aF(Ar, 64, m0, k0);
      v16h fAi  = load_aF(Ai, 64, m0, k0);
      v16h fAiN = load_aF(AiN, 64, m0, k0);
      v16h fBr  = load_bLDS(sbr, k0);
      v16h fBi  = load_bLDS(sbi, k0);
      or_ = wmma16(fAr, fBr, or_);
      or_ = wmma16(fAiN, fBi, or_);
      oi_ = wmma16(fAr, fBi, oi_);
      oi_ = wmma16(fAi, fBr, oi_);
    }
#pragma unroll
    for (int r = 0; r < 8; ++r) {
      int m = m0 + r + ((lane >> 4) << 3);
      size_t o = base + (size_t)m * LDFREQ + n0 + (lane & 15);
      Or[o] = (_Float16)or_[r];
      Oi[o] = (_Float16)oi_[r];
    }
  }
}

// ---------------------------------------------------------------------------
// Kernel 4: fused 2-layer block-diagonal complex MLP, in place on Z (f16).
// One wave = 16 rows x one 96-ch block; layer-1 ReLU'd into an LDS f16 strip,
// layer-2 A-fragments come from LDS (contiguous ds_load_b128s).
// ---------------------------------------------------------------------------
__global__ void __launch_bounds__(256) mlp_kernel(
    const _Float16* __restrict__ w1rT, const _Float16* __restrict__ w1iT,
    const _Float16* __restrict__ w1iNT,
    const _Float16* __restrict__ w2rT, const _Float16* __restrict__ w2iT,
    const _Float16* __restrict__ w2iNT,
    const float* __restrict__ b1, const float* __restrict__ b2,
    _Float16* __restrict__ Zr, _Float16* __restrict__ Zi) {
  __shared__ _Float16 s_r1[8][16 * BSZ];
  __shared__ _Float16 s_i1[8][16 * BSZ];
  int wvid = threadIdx.x >> 5;       // == block index nb
  int lane = threadIdx.x & 31;
  int nb = wvid;
  int m0 = blockIdx.x * 16;
  const _Float16* W1r  = w1rT  + (size_t)nb * BSZ * BSZ;
  const _Float16* W1i  = w1iT  + (size_t)nb * BSZ * BSZ;
  const _Float16* W1iN = w1iNT + (size_t)nb * BSZ * BSZ;
  const _Float16* W2r  = w2rT  + (size_t)nb * BSZ * BSZ;
  const _Float16* W2i  = w2iT  + (size_t)nb * BSZ * BSZ;
  const _Float16* W2iN = w2iNT + (size_t)nb * BSZ * BSZ;
  const _Float16* Xr = Zr + nb * BSZ;
  const _Float16* Xi = Zi + nb * BSZ;

  for (int ct = 0; ct < 6; ++ct) {
    int n0 = ct * 16;
    v8f r1, i1;
    float br = b1[(0 * NBLK + nb) * BSZ + n0 + (lane & 15)];
    float bi = b1[(1 * NBLK + nb) * BSZ + n0 + (lane & 15)];
#pragma unroll
    for (int r = 0; r < 8; ++r) { r1[r] = br; i1[r] = bi; }
    for (int k0 = 0; k0 < BSZ; k0 += 32) {
      v16h aXr = load_aF(Xr, DIMC, m0, k0);
      v16h aXi = load_aF(Xi, DIMC, m0, k0);
      v16h bR  = load_bT(W1r,  BSZ, k0, n0);
      v16h bI  = load_bT(W1i,  BSZ, k0, n0);
      v16h bIN = load_bT(W1iN, BSZ, k0, n0);
      r1 = wmma16(aXr, bR, r1);
      r1 = wmma16(aXi, bIN, r1);
      i1 = wmma16(aXr, bI, i1);
      i1 = wmma16(aXi, bR, i1);
    }
#pragma unroll
    for (int r = 0; r < 8; ++r) {
      int mm = r + ((lane >> 4) << 3);
      int nn = n0 + (lane & 15);
      s_r1[wvid][mm * BSZ + nn] = (_Float16)fmaxf(r1[r], 0.0f);
      s_i1[wvid][mm * BSZ + nn] = (_Float16)fmaxf(i1[r], 0.0f);
    }
  }

  const _Float16* sr = &s_r1[wvid][0];
  const _Float16* si = &s_i1[wvid][0];
  for (int ct = 0; ct < 6; ++ct) {
    int n0 = ct * 16;
    v8f r2, i2;
    float br = b2[(0 * NBLK + nb) * BSZ + n0 + (lane & 15)];
    float bi = b2[(1 * NBLK + nb) * BSZ + n0 + (lane & 15)];
#pragma unroll
    for (int r = 0; r < 8; ++r) { r2[r] = br; i2[r] = bi; }
    for (int k0 = 0; k0 < BSZ; k0 += 32) {
      v16h aR = load_aF(sr, BSZ, 0, k0);
      v16h aI = load_aF(si, BSZ, 0, k0);
      v16h bR  = load_bT(W2r,  BSZ, k0, n0);
      v16h bI  = load_bT(W2i,  BSZ, k0, n0);
      v16h bIN = load_bT(W2iN, BSZ, k0, n0);
      r2 = wmma16(aR, bR, r2);
      r2 = wmma16(aI, bIN, r2);
      i2 = wmma16(aR, bI, i2);
      i2 = wmma16(aI, bR, i2);
    }
#pragma unroll
    for (int r = 0; r < 8; ++r) {
      int m = m0 + r + ((lane >> 4) << 3);
      int nn = nb * BSZ + n0 + (lane & 15);
      Zr[(size_t)m * DIMC + nn] = (_Float16)sshrink(r2[r]);
      Zi[(size_t)m * DIMC + nn] = (_Float16)sshrink(i2[r]);
    }
  }
}

// ---------------------------------------------------------------------------
// Kernel 6: irfft along W + accumulate into d_out (bias already there).
// One workgroup per (bh, nt-octet); waves TDM-stage 33x16 Ur/Ui strips,
// K zero-guarded at 33 on the LDS read side (basis also zero-padded).
// ---------------------------------------------------------------------------
__global__ void __launch_bounds__(256) idft_w_add_kernel(
    const _Float16* __restrict__ DrT, const _Float16* __restrict__ DiT,
    const _Float16* __restrict__ Ur, const _Float16* __restrict__ Ui,
    float* __restrict__ out) {
  __shared__ _Float16 s_u[8][2][KWF * 16];   // ~16.9 KB
  int wvid = threadIdx.x >> 5;
  int lane = threadIdx.x & 31;
  int wg = blockIdx.x;
  int ng = wg % 6;
  int bh = wg / 6;                            // 0..511
  int n0 = (ng * 8 + wvid) * 16;
  size_t ibase = (size_t)bh * LDFREQ;
  size_t obase = (size_t)bh * WWW * DIMC;
#if HAVE_TDM
  uint32_t off_r = (uint32_t)(wvid * 2 * KWF * 16 * sizeof(_Float16));
  uint32_t off_i = off_r + KWF * 16 * sizeof(_Float16);
  tdm_load_tile_f16(off_r, Ur + ibase + n0, 16, KWF, DIMC);
  tdm_load_tile_f16(off_i, Ui + ibase + n0, 16, KWF, DIMC);
  tdm_wait();
#else
  manual_stage(&s_u[wvid][0][0], Ur + ibase + n0, KWF, DIMC);
  manual_stage(&s_u[wvid][1][0], Ui + ibase + n0, KWF, DIMC);
#endif
  const _Float16* sur = &s_u[wvid][0][0];
  const _Float16* sui = &s_u[wvid][1][0];
  for (int mt = 0; mt < 4; ++mt) {
    int m0 = mt * 16;
    v8f acc = {};
    for (int k0 = 0; k0 < 64; k0 += 32) {
      v16h fAr = load_aF(DrT, 64, m0, k0);
      v16h fAi = load_aF(DiT, 64, m0, k0);
      v16h fBr = load_bLDS_guard(sur, k0, KWF);
      v16h fBi = load_bLDS_guard(sui, k0, KWF);
      acc = wmma16(fAr, fBr, acc);
      acc = wmma16(fAi, fBi, acc);
    }
#pragma unroll
    for (int r = 0; r < 8; ++r) {
      int m = m0 + r + ((lane >> 4) << 3);   // w
      size_t o = obase + (size_t)m * DIMC + n0 + (lane & 15);
      out[o] += acc[r];
    }
  }
}

// ---------------------------------------------------------------------------
extern "C" void kernel_launch(void* const* d_in, const int* in_sizes, int n_in,
                              void* d_out, int out_size, void* d_ws, size_t ws_size,
                              hipStream_t stream) {
  (void)in_sizes; (void)n_in; (void)out_size; (void)ws_size;
  const float* x  = (const float*)d_in[0];
  const float* w1 = (const float*)d_in[1];
  const float* b1 = (const float*)d_in[2];
  const float* w2 = (const float*)d_in[3];
  const float* b2 = (const float*)d_in[4];
  const float* Wb = (const float*)d_in[5];
  const float* bb = (const float*)d_in[6];
  float* out = (float*)d_out;

  uint8_t* p = (uint8_t*)d_ws;
  auto carve = [&](size_t bytes) -> void* {
    void* r = (void*)p;
    p += (bytes + 255) & ~(size_t)255;
    return r;
  };
  const size_t XELEMS = (size_t)BATCH * HH * WWW * DIMC;   // 25,165,824
  const size_t FREQ_ELEMS = (size_t)NPOS * DIMC;           // 12,976,128
  _Float16* x_h = (_Float16*)carve(XELEMS * 2);
  _Float16* Yr = (_Float16*)carve(FREQ_ELEMS * 2);
  _Float16* Yi = (_Float16*)carve(FREQ_ELEMS * 2);
  _Float16* Zr = (_Float16*)carve(FREQ_ELEMS * 2);
  _Float16* Zi = (_Float16*)carve(FREQ_ELEMS * 2);
  _Float16* Wb_h = (_Float16*)carve((size_t)DIMC * DIMC * 2);
  const size_t WSZB = (size_t)NBLK * BSZ * BSZ * 2;
  _Float16* w1rT  = (_Float16*)carve(WSZB);
  _Float16* w1iT  = (_Float16*)carve(WSZB);
  _Float16* w1iNT = (_Float16*)carve(WSZB);
  _Float16* w2rT  = (_Float16*)carve(WSZB);
  _Float16* w2iT  = (_Float16*)carve(WSZB);
  _Float16* w2iNT = (_Float16*)carve(WSZB);
  _Float16* CwR  = (_Float16*)carve(48 * 64 * 2);
  _Float16* CwI  = (_Float16*)carve(48 * 64 * 2);
  _Float16* ChR  = (_Float16*)carve(64 * 64 * 2);
  _Float16* ChI  = (_Float16*)carve(64 * 64 * 2);
  _Float16* ChIN = (_Float16*)carve(64 * 64 * 2);
  _Float16* iChR = (_Float16*)carve(64 * 64 * 2);
  _Float16* iChI = (_Float16*)carve(64 * 64 * 2);
  _Float16* iChIN= (_Float16*)carve(64 * 64 * 2);
  _Float16* DrT  = (_Float16*)carve(64 * 64 * 2);
  _Float16* DiT  = (_Float16*)carve(64 * 64 * 2);

  // A) x -> f16 (float4 vectorized): 6,291,456 quads
  cvt_x_kernel<<<(int)(XELEMS / 4 / 256), 256, 0, stream>>>(x, x_h);

  // 0) weight/basis prep: 896000 threads
  prep_kernel<<<3500, 256, 0, stream>>>(w1, w2, Wb, Wb_h,
      w1rT, w1iT, w1iNT, w2rT, w2iT, w2iNT,
      CwR, CwI, ChR, ChI, ChIN, iChR, iChI, iChIN, DrT, DiT);

  // 1) bias GEMM (overwrites d_out): 2048*48 wave-tiles
  bias_gemm_kernel<<<(2048 * 48) / 8, 256, 0, stream>>>(x_h, Wb_h, bb, out);

  // 2) rfft-W, TDM-staged X strips: 512*6 workgroups
  dft_w_kernel<<<512 * 6, 256, 0, stream>>>(x_h, CwR, CwI, Yr, Yi);

  // 3) fft-H (Y -> Z), TDM-staged B: 8*33*6 workgroups
  dft_h_kernel<<<8 * KWF * 6, 256, 0, stream>>>(ChR, ChI, ChIN, Yr, Yi, Zr, Zi);

  // 4) block-diagonal complex MLP, in place on Z
  mlp_kernel<<<1056, 256, 0, stream>>>(w1rT, w1iT, w1iNT, w2rT, w2iT, w2iNT,
                                       b1, b2, Zr, Zi);

  // 5) ifft-H (Z -> Y), TDM-staged B
  dft_h_kernel<<<8 * KWF * 6, 256, 0, stream>>>(iChR, iChI, iChIN, Zr, Zi, Yr, Yi);

  // 6) irfft-W + accumulate into d_out: 512*6 workgroups
  idft_w_add_kernel<<<512 * 6, 256, 0, stream>>>(DrT, DiT, Yr, Yi, out);
}
